// RN_5488968204646
// MI455X (gfx1250) — matrix operand order repaired and checked by reading, not compiled, not run on hardware
//
#include <hip/hip_runtime.h>
#include <hip/hip_bf16.h>

typedef __attribute__((ext_vector_type(8)))  _Float16 v8h;
typedef __attribute__((ext_vector_type(16))) _Float16 v16h;
typedef __attribute__((ext_vector_type(8)))  float    v8f;

union F16Frag { v16h v; v8h h[2]; };

#define BN_EPS 1e-5f

// ---------------------------------------------------------------------------
// Conv 3x3 stride 2 pad 1, +bias, ReLU (Cout fixed at 24)
// ---------------------------------------------------------------------------
__global__ void conv2d_s2_relu(const float* __restrict__ x, const float* __restrict__ w,
                               const float* __restrict__ bias, float* __restrict__ y,
                               int N, int Cin, int H, int W, int Ho, int Wo) {
  int idx = blockIdx.x * blockDim.x + threadIdx.x;
  int total = N * 24 * Ho * Wo;
  if (idx >= total) return;
  int wo = idx % Wo; int t = idx / Wo;
  int ho = t % Ho;   t /= Ho;
  int co = t % 24;   int n = t / 24;
  float s = bias[co];
  int hi0 = ho * 2 - 1, wi0 = wo * 2 - 1;
  for (int ci = 0; ci < Cin; ++ci) {
    const float* xp = x + (size_t)(n * Cin + ci) * H * W;
    const float* wp = w + (co * Cin + ci) * 9;
#pragma unroll
    for (int kh = 0; kh < 3; ++kh) {
      int hi = hi0 + kh;
      if ((unsigned)hi >= (unsigned)H) continue;
#pragma unroll
      for (int kw = 0; kw < 3; ++kw) {
        int wi = wi0 + kw;
        if ((unsigned)wi >= (unsigned)W) continue;
        s += xp[hi * W + wi] * wp[kh * 3 + kw];
      }
    }
  }
  y[idx] = fmaxf(s, 0.f);
}

// ---------------------------------------------------------------------------
// BatchNorm (training-mode batch stats): per-(n,c) plane partial reduce
// stats[c] = sum, stats[32+c] = sumsq   (24 channels)
// ---------------------------------------------------------------------------
__global__ void bn_stats(const float* __restrict__ y, float* __restrict__ stats, int HW) {
  int plane = blockIdx.x;            // n*24 + c
  int c = plane % 24;
  const float* p = y + (size_t)plane * HW;
  float s = 0.f, s2 = 0.f;
  for (int i = threadIdx.x; i < HW; i += blockDim.x) { float v = p[i]; s += v; s2 += v * v; }
  __shared__ float red0[256];
  __shared__ float red1[256];
  red0[threadIdx.x] = s; red1[threadIdx.x] = s2;
  __syncthreads();
  for (int offd = 128; offd; offd >>= 1) {
    if (threadIdx.x < offd) {
      red0[threadIdx.x] += red0[threadIdx.x + offd];
      red1[threadIdx.x] += red1[threadIdx.x + offd];
    }
    __syncthreads();
  }
  if (threadIdx.x == 0) {
    atomicAdd(&stats[c],      red0[0]);
    atomicAdd(&stats[32 + c], red1[0]);
  }
}

__global__ void bn_apply(float* __restrict__ y, const float* __restrict__ stats,
                         const float* __restrict__ g, const float* __restrict__ be,
                         int HW, int total) {
  int idx = blockIdx.x * blockDim.x + threadIdx.x;
  if (idx >= total) return;
  int c = (idx / HW) % 24;
  float cnt = (float)(512 * HW);
  float mu = stats[c] / cnt;
  float var = stats[32 + c] / cnt - mu * mu;
  y[idx] = (y[idx] - mu) * rsqrtf(var + BN_EPS) * g[c] + be[c];
}

// ---------------------------------------------------------------------------
// Weight conversion to f16 (+ zero-pad g_w1 from [256][63] to [256][64])
// ---------------------------------------------------------------------------
__global__ void f32_to_f16(const float* __restrict__ src, _Float16* __restrict__ dst, int n) {
  int i = blockIdx.x * blockDim.x + threadIdx.x;
  if (i < n) dst[i] = (_Float16)src[i];
}

__global__ void pad_w1(const float* __restrict__ src, _Float16* __restrict__ dst) {
  int idx = blockIdx.x * blockDim.x + threadIdx.x;       // 256*64
  if (idx >= 16384) return;
  int c = idx & 63, r = idx >> 6;
  dst[idx] = (c < 63) ? (_Float16)src[r * 63 + c] : (_Float16)0.f;
}

// ---------------------------------------------------------------------------
// Build pair features: xp[row = b*625 + j*25 + p][64] =
//   [x_flat[b,p](24), coord(p)(2), x_flat[b,j](24), coord(j)(2), qst[b](11), 0]
// ---------------------------------------------------------------------------
__global__ void pair_build(const float* __restrict__ x4, const float* __restrict__ qst,
                           _Float16* __restrict__ xp) {
  int idx = blockIdx.x * blockDim.x + threadIdx.x;       // 320000*64
  if (idx >= 320000 * 64) return;
  int col = idx & 63; int row = idx >> 6;
  int p = row % 25; int t = row / 25; int j = t % 25; int b = t / 25;
  float v;
  if (col < 24)        v = x4[(b * 24 + col) * 25 + p];
  else if (col == 24)  v = (float)(p / 5 - 2) * 0.5f;
  else if (col == 25)  v = (float)(p % 5 - 2) * 0.5f;
  else if (col < 50)   v = x4[(b * 24 + (col - 26)) * 25 + j];
  else if (col == 50)  v = (float)(j / 5 - 2) * 0.5f;
  else if (col == 51)  v = (float)(j % 5 - 2) * 0.5f;
  else if (col < 63)   v = qst[b * 11 + (col - 52)];
  else                 v = 0.f;
  xp[idx] = (_Float16)v;
}

// ---------------------------------------------------------------------------
// Async copy of one weight K-chunk (256 rows x 64 cols f16 = 32KB) into LDS.
// 256 threads x 8 iterations of global_load_async_to_lds_b128 (ASYNCcnt +8/wave).
// ---------------------------------------------------------------------------
__device__ __forceinline__ void async_copy_wchunk(_Float16* dstLds,
                                                  const _Float16* __restrict__ src,
                                                  int K, int kc, int tid) {
  unsigned ldsBase = (unsigned)(size_t)dstLds;   // low 32 bits = LDS byte offset
  for (int c = tid; c < 2048; c += 256) {        // 8 uniform iterations
    int n = c >> 3, cc = c & 7;
    unsigned lofs = ldsBase + (unsigned)((n * 64 + cc * 8) * 2);
    unsigned long long g = (unsigned long long)(size_t)(src + (size_t)n * K + kc + cc * 8);
    asm volatile("global_load_async_to_lds_b128 %0, %1, off"
                 :: "v"(lofs), "v"(g) : "memory");
  }
}

// ---------------------------------------------------------------------------
// Fused g-MLP: 4 layers (64->256, 256->256 x3), activations LDS-resident.
// Block = 256 threads (8 waves), 128 rows/block. Each wave owns a 32-row band
// and half of N (8 n-tiles): 16 accumulators, each B fragment feeds 2 WMMAs.
// Weight K-chunks double-buffered in LDS via async loads. Final layer does a
// per-column segmented reduction in LDS, then <=2 atomics per column.
// ---------------------------------------------------------------------------
#define GTM 128
__global__ __launch_bounds__(256) void g_mlp_fused(
    const _Float16* __restrict__ xp,   // [320000][64]
    const _Float16* __restrict__ wg,   // w1p[256][64] | w2,w3,w4 [256][256]
    const float* __restrict__ gb1, const float* __restrict__ gb2,
    const float* __restrict__ gb3, const float* __restrict__ gb4,
    float* __restrict__ xg) {          // [512][256] (pre-zeroed)
  extern __shared__ _Float16 smem[];
  _Float16* act  = smem;                     // [128][256] f16 = 64KB
  _Float16* wch0 = smem + GTM * 256;         // [256][64]  f16 = 32KB
  _Float16* wch1 = wch0 + 256 * 64;          // [256][64]  f16 = 32KB
  const int tid  = threadIdx.x;
  const int wave = tid >> 5;
  const int lane = tid & 31;
  const int half = lane >> 4;
  const int m    = lane & 15;
  const int band = wave >> 1;                // 0..3 -> 32-row band
  const int nh   = (wave & 1) * 8;           // n-tile base (0 or 8)
  const long rowBase = (long)blockIdx.x * GTM;
  const int rw0 = band * 32;
  const int rw1 = rw0 + 16;

  // Stage this tile's pair features into act (cols 0..63, row stride 256)
  {
    const v8h* src = (const v8h*)(xp + rowBase * 64);
    for (int c = tid; c < 1024; c += 256) {        // 128 rows * 8 chunks
      int r = c >> 3, cc = c & 7;
      *(v8h*)(act + r * 256 + cc * 8) = src[c];
    }
  }
  __syncthreads();

  const v8f vzero = {};
  v8f acc[16];                                // [sub-band 2][n-tile 8]

  for (int L = 0; L < 4; ++L) {
#pragma unroll
    for (int q = 0; q < 16; ++q) acc[q] = vzero;
    const int K = (L == 0) ? 64 : 256;
    const int nchunks = K >> 6;
    const _Float16* wsrc = wg + ((L == 0) ? 0 : (16384 + (L - 1) * 65536));
    const float* bias = (L == 0) ? gb1 : (L == 1) ? gb2 : (L == 2) ? gb3 : gb4;

    async_copy_wchunk(wch0, wsrc, K, 0, tid);                 // prime chunk 0

    for (int ci = 0; ci < nchunks; ++ci) {
      _Float16* cur = (ci & 1) ? wch1 : wch0;
      if (ci + 1 < nchunks) {
        async_copy_wchunk((ci & 1) ? wch0 : wch1, wsrc, K, (ci + 1) * 64, tid);
        asm volatile("s_wait_asynccnt 0x8" ::: "memory");     // chunk ci landed
      } else {
        asm volatile("s_wait_asynccnt 0x0" ::: "memory");
      }
      __syncthreads();                                        // publish chunk ci

#pragma unroll
      for (int kk = 0; kk < 2; ++kk) {                        // two K=32 steps
        const int k0 = ci * 64 + kk * 32;
        const _Float16* ar0 = act + (rw0 + m) * 256;
        const _Float16* ar1 = act + (rw1 + m) * 256;
        F16Frag A0, A1;
        A0.h[0] = *(const v8h*)(ar0 + k0 + half * 16);
        A0.h[1] = *(const v8h*)(ar0 + k0 + half * 16 + 8);
        A1.h[0] = *(const v8h*)(ar1 + k0 + half * 16);
        A1.h[1] = *(const v8h*)(ar1 + k0 + half * 16 + 8);
#pragma unroll
        for (int n = 0; n < 8; ++n) {
          const _Float16* brow = cur + ((nh + n) * 16 + m) * 64 + kk * 32 + half * 16;
          F16Frag B;
          B.h[0] = *(const v8h*)(brow);
          B.h[1] = *(const v8h*)(brow + 8);
          acc[n] = __builtin_amdgcn_wmma_f32_16x16x32_f16(
              false, A0.v, false, B.v, (short)0, acc[n], false, false);
          acc[8 + n] = __builtin_amdgcn_wmma_f32_16x16x32_f16(
              false, A1.v, false, B.v, (short)0, acc[8 + n], false, false);
        }
      }
      __syncthreads();     // all waves done with chunk ci before buffer reuse
    }

    if (L < 3) {
      // bias + relu -> f16 back into act (wave writes only its own region)
#pragma unroll
      for (int s = 0; s < 2; ++s) {
        const int rw = s ? rw1 : rw0;
#pragma unroll
        for (int n = 0; n < 8; ++n) {
          const int col = (nh + n) * 16 + m;
          float bn = bias[col];
#pragma unroll
          for (int r = 0; r < 8; ++r) {
            float v = fmaxf(acc[s * 8 + n][r] + bn, 0.f);
            act[(rw + r + half * 8) * 256 + col] = (_Float16)v;
          }
        }
      }
    } else {
      // final layer: bias+relu -> f32 LDS buffer (reuse all 128KB), then
      // per-column segmented sum over the 128 rows (<=2 images per tile)
      float* fbuf = (float*)smem;               // [128][256] f32 = 128KB
#pragma unroll
      for (int s = 0; s < 2; ++s) {
        const int rw = s ? rw1 : rw0;
#pragma unroll
        for (int n = 0; n < 8; ++n) {
          const int col = (nh + n) * 16 + m;
          float bn = bias[col];
#pragma unroll
          for (int r = 0; r < 8; ++r) {
            float v = fmaxf(acc[s * 8 + n][r] + bn, 0.f);
            fbuf[(rw + r + half * 8) * 256 + col] = v;
          }
        }
      }
      __syncthreads();
      const int col = tid;                      // 256 threads x 256 cols
      float s = 0.f;
      long prevb = rowBase / 625;
      for (int r = 0; r < GTM; ++r) {
        long b = (rowBase + r) / 625;
        if (b != prevb) {
          atomicAdd(&xg[prevb * 256 + col], s);
          s = 0.f;
          prevb = b;
        }
        s += fbuf[r * 256 + col];
      }
      atomicAdd(&xg[prevb * 256 + col], s);
    }
  }
}

// ---------------------------------------------------------------------------
// One-wave WMMA GEMM tile: Y[16x16] = relu(X[16xK] * W[N][K]^T + bias)
// grid.x = (M/16)*16 ; N fixed at 256.
// ---------------------------------------------------------------------------
__global__ __launch_bounds__(32) void wmma_linear_relu(
    const _Float16* __restrict__ X, const _Float16* __restrict__ Wt,
    const float* __restrict__ bias, _Float16* __restrict__ Y, int K) {
  const int nt = blockIdx.x & 15;
  const int mt = blockIdx.x >> 4;
  const int lane = threadIdx.x;
  const int half = lane >> 4;
  const int m = lane & 15;
  v8f acc = {};
  const _Float16* arow = X  + (size_t)(mt * 16 + m) * K;
  const _Float16* brow = Wt + (size_t)(nt * 16 + m) * K;
  for (int k0 = 0; k0 < K; k0 += 32) {
    F16Frag A, B;
    A.h[0] = *(const v8h*)(arow + k0 + half * 16);
    A.h[1] = *(const v8h*)(arow + k0 + half * 16 + 8);
    B.h[0] = *(const v8h*)(brow + k0 + half * 16);
    B.h[1] = *(const v8h*)(brow + k0 + half * 16 + 8);
    acc = __builtin_amdgcn_wmma_f32_16x16x32_f16(
        false, A.v, false, B.v, (short)0, acc, false, false);
  }
  float bn = bias[nt * 16 + m];
#pragma unroll
  for (int r = 0; r < 8; ++r) {
    float v = fmaxf(acc[r] + bn, 0.f);
    Y[(size_t)(mt * 16 + r + half * 8) * 256 + nt * 16 + m] = (_Float16)v;
  }
}

// ---------------------------------------------------------------------------
// fc3 (256->10) + log_softmax, one thread per image
// ---------------------------------------------------------------------------
__global__ void head_logsoftmax(const _Float16* __restrict__ xf, const float* __restrict__ w,
                                const float* __restrict__ b, float* __restrict__ out) {
  int i = blockIdx.x * blockDim.x + threadIdx.x;
  if (i >= 512) return;
  float l[10];
  float mx = -1e30f;
#pragma unroll
  for (int o = 0; o < 10; ++o) {
    float s = b[o];
    for (int k = 0; k < 256; ++k) s += (float)xf[i * 256 + k] * w[o * 256 + k];
    l[o] = s;
    mx = fmaxf(mx, s);
  }
  float se = 0.f;
#pragma unroll
  for (int o = 0; o < 10; ++o) se += expf(l[o] - mx);
  float lse = logf(se);
#pragma unroll
  for (int o = 0; o < 10; ++o) out[i * 10 + o] = l[o] - mx - lse;
}

// ---------------------------------------------------------------------------
extern "C" void kernel_launch(void* const* d_in, const int* in_sizes, int n_in,
                              void* d_out, int out_size, void* d_ws, size_t ws_size,
                              hipStream_t stream) {
  (void)in_sizes; (void)n_in; (void)out_size; (void)ws_size;
  const float* img = (const float*)d_in[0];
  const float* qst = (const float*)d_in[1];
  const float* cw[4] = {(const float*)d_in[2],  (const float*)d_in[6],
                        (const float*)d_in[10], (const float*)d_in[14]};
  const float* cb[4] = {(const float*)d_in[3],  (const float*)d_in[7],
                        (const float*)d_in[11], (const float*)d_in[15]};
  const float* bg[4] = {(const float*)d_in[4],  (const float*)d_in[8],
                        (const float*)d_in[12], (const float*)d_in[16]};
  const float* bb[4] = {(const float*)d_in[5],  (const float*)d_in[9],
                        (const float*)d_in[13], (const float*)d_in[17]};
  const float* g_w1 = (const float*)d_in[18]; const float* g_b1 = (const float*)d_in[19];
  const float* g_w2 = (const float*)d_in[20]; const float* g_b2 = (const float*)d_in[21];
  const float* g_w3 = (const float*)d_in[22]; const float* g_b3 = (const float*)d_in[23];
  const float* g_w4 = (const float*)d_in[24]; const float* g_b4 = (const float*)d_in[25];
  const float* f_w1 = (const float*)d_in[26]; const float* f_b1 = (const float*)d_in[27];
  const float* fc2_w = (const float*)d_in[28]; const float* fc2_b = (const float*)d_in[29];
  const float* fc3_w = (const float*)d_in[30]; const float* fc3_b = (const float*)d_in[31];
  float* out = (float*)d_out;

  // workspace carve-up
  char* ws = (char*)d_ws;
  size_t off = 0;
  auto take = [&](size_t bytes) {
    char* p = ws + off;
    off = (off + bytes + 255) & ~(size_t)255;
    return p;
  };
  float*    x1    = (float*)take(17743872ULL * 4);   // [512,24,38,38]
  float*    x2    = (float*)take(4435968ULL * 4);    // [512,24,19,19]
  float*    x3    = (float*)take(1228800ULL * 4);    // [512,24,10,10]
  float*    x4    = (float*)take(307200ULL * 4);     // [512,24,5,5]
  float*    stats = (float*)take(64 * 4);
  float*    xg    = (float*)take(131072ULL * 4);     // [512,256]
  _Float16* xp    = (_Float16*)take(20480000ULL * 2);// [320000,64]
  _Float16* wg    = (_Float16*)take(212992ULL * 2);  // g weights f16
  _Float16* fw    = (_Float16*)take(131072ULL * 2);  // f_w1 + fc2_w f16
  _Float16* xgh   = (_Float16*)take(131072ULL * 2);
  _Float16* xf1   = (_Float16*)take(131072ULL * 2);
  _Float16* xf2   = (_Float16*)take(131072ULL * 2);

  const int N = 512;
  struct Lyr { const float* in; float* outb; int Cin, H, W, Ho, Wo; };
  Lyr L[4] = {
    { img, x1, 3, 75, 75, 38, 38 },
    { x1,  x2, 24, 38, 38, 19, 19 },
    { x2,  x3, 24, 19, 19, 10, 10 },
    { x3,  x4, 24, 10, 10, 5, 5 },
  };
  for (int l = 0; l < 4; ++l) {
    int total = N * 24 * L[l].Ho * L[l].Wo;
    conv2d_s2_relu<<<(total + 255) / 256, 256, 0, stream>>>(
        L[l].in, cw[l], cb[l], L[l].outb, N, L[l].Cin, L[l].H, L[l].W, L[l].Ho, L[l].Wo);
    hipMemsetAsync(stats, 0, 64 * 4, stream);
    int HW = L[l].Ho * L[l].Wo;
    bn_stats<<<N * 24, 256, 0, stream>>>(L[l].outb, stats, HW);
    bn_apply<<<(total + 255) / 256, 256, 0, stream>>>(L[l].outb, stats, bg[l], bb[l], HW, total);
  }

  // weights -> f16 (g_w1 padded K 63->64)
  pad_w1<<<64, 256, 0, stream>>>(g_w1, wg);
  f32_to_f16<<<256, 256, 0, stream>>>(g_w2, wg + 16384, 65536);
  f32_to_f16<<<256, 256, 0, stream>>>(g_w3, wg + 81920, 65536);
  f32_to_f16<<<256, 256, 0, stream>>>(g_w4, wg + 147456, 65536);
  f32_to_f16<<<256, 256, 0, stream>>>(f_w1, fw, 65536);
  f32_to_f16<<<256, 256, 0, stream>>>(fc2_w, fw + 65536, 65536);

  // pair features (f16, K padded to 64)
  pair_build<<<80000, 256, 0, stream>>>(x4, qst, xp);

  // fused g-MLP + pair-sum (128KB dynamic LDS: act + double-buffered weights)
  hipMemsetAsync(xg, 0, 131072ULL * 4, stream);
  g_mlp_fused<<<2500, 256, 131072, stream>>>(xp, wg, g_b1, g_b2, g_b3, g_b4, xg);

  // f-MLP (WMMA) + head
  f32_to_f16<<<512, 256, 0, stream>>>(xg, xgh, 131072);
  wmma_linear_relu<<<512, 32, 0, stream>>>(xgh, fw, f_b1, xf1, 256);
  wmma_linear_relu<<<512, 32, 0, stream>>>(xf1, fw + 65536, fc2_b, xf2, 256);
  head_logsoftmax<<<2, 256, 0, stream>>>(xf2, fc3_w, fc3_b, out);
}
